// SparsegenAttentionEntityPooler_75333726371898
// MI455X (gfx1250) — compile-verified, matrix operand order checked by reading, not run on hardware
//
#include <hip/hip_runtime.h>

#define B_ 64
#define S_ 2048
#define H_ 768
#define NEGV (-1e30f)

typedef __attribute__((ext_vector_type(16))) _Float16 v16h;
typedef __attribute__((ext_vector_type(8)))  float    v8f;

// ---------------------------------------------------------------------------
// Kernel 0: dotA[b] = dot(pooled_tokens[b,:], w_align[:H])
// ---------------------------------------------------------------------------
__global__ void k_dot_pooled(const float* __restrict__ pooled,
                             const float* __restrict__ w_align,
                             float* __restrict__ dotA) {
  __shared__ float red[256];
  const int b = blockIdx.x, t = threadIdx.x;
  float s = 0.f;
  for (int i = t; i < H_; i += 256)
    s = fmaf(pooled[(size_t)b * H_ + i], w_align[i], s);
  red[t] = s;
  __syncthreads();
  for (int off = 128; off > 0; off >>= 1) {
    if (t < off) red[t] += red[t + off];
    __syncthreads();
  }
  if (t == 0) dotA[b] = red[0];
}

// ---------------------------------------------------------------------------
// Kernel 1: scores via WMMA. One wave computes 16 consecutive s-rows:
//   scores[b,s] = mask[b,s]*(dotA[b] + dot(hidden[b,s,:], w2)) + bias
// A = 16x32 f16 tile of hidden; B = w2 chunk replicated across 16 columns.
// K loop unrolled x4 so the scheduler can clause the global/ds loads ahead
// of 4 back-to-back v_wmma ops (HBM latency hiding; kernel is BW-bound).
// ---------------------------------------------------------------------------
__global__ void k_scores_wmma(const float* __restrict__ hidden,
                              const float* __restrict__ token_mask,
                              const float* __restrict__ w_align,
                              const float* __restrict__ b_align,
                              const float* __restrict__ dotA,
                              float* __restrict__ scores) {
  __shared__ _Float16 w2h[H_];
  const int t = threadIdx.x;
  for (int i = t; i < H_; i += 256) w2h[i] = (_Float16)w_align[H_ + i];
  __syncthreads();

  const int wave  = t >> 5;
  const int lane  = t & 31;
  const int tile  = blockIdx.x * 8 + wave;     // 16 s-rows per tile
  const int b     = tile / (S_ / 16);
  const int srow  = (tile % (S_ / 16)) * 16;
  const int m     = lane & 15;                 // A-matrix row within tile
  const int kbase = (lane >> 4) << 3;          // K sub-block per lane half

  const float* rowp = hidden + ((size_t)b * S_ + srow + m) * (size_t)H_;

  v8f acc = {};
#pragma unroll 4
  for (int kk = 0; kk < H_; kk += 32) {
    // Speculative prefetch (silently dropped if past mapping) — no branch.
    __builtin_prefetch(rowp + kk + 128, 0, 1);
    v16h a, bm;
#pragma unroll
    for (int j = 0; j < 8; ++j) {
      a[j]     = (_Float16)rowp[kk + kbase + j];        // K = base + j
      a[j + 8] = (_Float16)rowp[kk + 16 + kbase + j];   // K = 16 + base + j
      bm[j]     = w2h[kk + kbase + j];                  // B replicated over N
      bm[j + 8] = w2h[kk + 16 + kbase + j];
    }
    acc = __builtin_amdgcn_wmma_f32_16x16x32_f16(
        /*neg_a=*/false, a, /*neg_b=*/false, bm,
        /*c_mod=*/(short)0, acc, /*reuse_a=*/false, /*reuse_b=*/false);
  }

  const float bias = b_align[0];
  const float da   = dotA[b];
  if (m == 0) {                                // lanes 0 (rows 0-7), 16 (rows 8-15)
    const int rbase = (lane >> 4) << 3;
#pragma unroll
    for (int r = 0; r < 8; ++r) {
      const int s = srow + rbase + r;
      const float mk = token_mask[((size_t)b * S_ + s) * (size_t)H_];
      scores[(size_t)b * S_ + s] = mk * (da + acc[r]) + bias;
    }
  }
}

// ---------------------------------------------------------------------------
// Kernel 2: in-place sparsemax (sparsegen-lin, lam=0) per batch row.
// probs_io holds scores on entry, probabilities on exit.
// ---------------------------------------------------------------------------
__global__ void k_sparsemax(const float* __restrict__ token_mask,
                            float* __restrict__ probs_io) {
  __shared__ float z[S_];
  __shared__ float zs[S_];
  __shared__ float cs[S_];
  __shared__ unsigned char vf[S_];
  __shared__ float redf[256];
  __shared__ int   redi[256];
  __shared__ float f_zmax, f_tau;
  __shared__ int   i_nv;

  const int b = blockIdx.x, t = threadIdx.x;
  const int base = t * 8;

  // Load scores + validity; reduce max-over-valid and valid count.
  float lmax = -3.4e38f;
  int lcnt = 0;
#pragma unroll
  for (int j = 0; j < 8; ++j) {
    const int s = base + j;
    const float sc = probs_io[(size_t)b * S_ + s];
    const unsigned char v = (token_mask[((size_t)b * S_ + s) * (size_t)H_] > 0.f);
    vf[s] = v;
    z[s]  = sc;
    if (v) { lmax = fmaxf(lmax, sc); lcnt++; }
  }
  redf[t] = lmax; redi[t] = lcnt;
  __syncthreads();
  for (int off = 128; off > 0; off >>= 1) {
    if (t < off) {
      redf[t] = fmaxf(redf[t], redf[t + off]);
      redi[t] += redi[t + off];
    }
    __syncthreads();
  }
  if (t == 0) { f_zmax = redf[0]; i_nv = redi[0]; }
  __syncthreads();
  const float zmax = f_zmax;
  const int nv = i_nv;

#pragma unroll
  for (int j = 0; j < 8; ++j) {
    const int s = base + j;
    const float zz = z[s] - zmax;
    z[s]  = zz;
    zs[s] = vf[s] ? zz : NEGV;
  }
  __syncthreads();

  // Bitonic sort of zs, descending (valid entries first).
  for (int k = 2; k <= S_; k <<= 1) {
    for (int j = k >> 1; j > 0; j >>= 1) {
      for (int i = t; i < S_; i += 256) {
        const int ixj = i ^ j;
        if (ixj > i) {
          const float a = zs[i], c = zs[ixj];
          const bool descRegion = ((i & k) == 0);
          if (descRegion ? (a < c) : (a > c)) { zs[i] = c; zs[ixj] = a; }
        }
      }
      __syncthreads();
    }
  }

  // zs_m = sorted_valid ? zs : 0, inclusive cumsum into cs (chunked scan).
  float loc[8];
  float run = 0.f;
#pragma unroll
  for (int j = 0; j < 8; ++j) {
    const int s = base + j;
    const float v = (s < nv) ? zs[s] : 0.f;
    zs[s] = v;
    run += v;
    loc[j] = run;
  }
  redf[t] = run;
  __syncthreads();
  for (int off = 1; off < 256; off <<= 1) {
    const float v = redf[t];
    const float add = (t >= off) ? redf[t - off] : 0.f;
    __syncthreads();
    redf[t] = v + add;
    __syncthreads();
  }
  const float prev = (t > 0) ? redf[t - 1] : 0.f;
#pragma unroll
  for (int j = 0; j < 8; ++j) cs[base + j] = loc[j] + prev;
  __syncthreads();

  // k = max rank with (1 + rank*zs_m > csum); tau = (sum zs_m over is_gt - 1)/k
  int lk = 0;
  float lsum = 0.f;
#pragma unroll
  for (int j = 0; j < 8; ++j) {
    const int s = base + j;
    const float zm = zs[s];
    const bool gt = (s < nv) && ((1.0f + (float)(s + 1) * zm) > cs[s]);
    if (gt) { lk = (s + 1 > lk) ? s + 1 : lk; lsum += zm; }
  }
  redi[t] = lk; redf[t] = lsum;
  __syncthreads();
  for (int off = 128; off > 0; off >>= 1) {
    if (t < off) {
      redi[t] = (redi[t] > redi[t + off]) ? redi[t] : redi[t + off];
      redf[t] += redf[t + off];
    }
    __syncthreads();
  }
  if (t == 0) {
    const int kk = (redi[0] > 1) ? redi[0] : 1;
    f_tau = (redf[0] - 1.0f) / (float)kk;
  }
  __syncthreads();
  const float tau = f_tau;

#pragma unroll
  for (int j = 0; j < 8; ++j) {
    const int s = base + j;
    const float p = vf[s] ? fmaxf(z[s] - tau, 0.f) : 0.f;
    probs_io[(size_t)b * S_ + s] = p;
  }
}

// ---------------------------------------------------------------------------
// Kernel 3: pooled[b,h] = sum_s probs[b,s]*hidden[b,s,h], skipping prob==0
// via deterministic prefix-sum compaction (no atomics -> replay-stable).
// grid = (B, H/256); each thread owns one h. Inner loop unrolled x4 so four
// support-row loads are in flight together.
// ---------------------------------------------------------------------------
__global__ void k_pool(const float* __restrict__ hidden,
                       const float* __restrict__ probs,
                       float* __restrict__ pooled) {
  __shared__ int   sidx[S_];
  __shared__ float spv[S_];
  __shared__ int   cnts[256];

  const int b = blockIdx.x, t = threadIdx.x;
  const int base = t * 8;

  float pv[8];
  int cnt = 0;
#pragma unroll
  for (int j = 0; j < 8; ++j) {
    pv[j] = probs[(size_t)b * S_ + base + j];
    if (pv[j] > 0.f) cnt++;
  }
  cnts[t] = cnt;
  __syncthreads();
  for (int off = 1; off < 256; off <<= 1) {
    const int v = cnts[t];
    const int add = (t >= off) ? cnts[t - off] : 0;
    __syncthreads();
    cnts[t] = v + add;
    __syncthreads();
  }
  const int total = cnts[255];
  int pos = cnts[t] - cnt;
#pragma unroll
  for (int j = 0; j < 8; ++j) {
    if (pv[j] > 0.f) { sidx[pos] = base + j; spv[pos] = pv[j]; pos++; }
  }
  __syncthreads();

  const int h = blockIdx.y * 256 + t;
  const float* __restrict__ hb = hidden + (size_t)b * S_ * H_ + h;
  float acc0 = 0.f, acc1 = 0.f, acc2 = 0.f, acc3 = 0.f;
  int i = 0;
  for (; i + 4 <= total; i += 4) {
    acc0 = fmaf(spv[i + 0], hb[(size_t)sidx[i + 0] * H_], acc0);
    acc1 = fmaf(spv[i + 1], hb[(size_t)sidx[i + 1] * H_], acc1);
    acc2 = fmaf(spv[i + 2], hb[(size_t)sidx[i + 2] * H_], acc2);
    acc3 = fmaf(spv[i + 3], hb[(size_t)sidx[i + 3] * H_], acc3);
  }
  for (; i < total; ++i)
    acc0 = fmaf(spv[i], hb[(size_t)sidx[i] * H_], acc0);
  pooled[(size_t)b * H_ + h] = ((acc0 + acc1) + (acc2 + acc3));
}

// ---------------------------------------------------------------------------
extern "C" void kernel_launch(void* const* d_in, const int* in_sizes, int n_in,
                              void* d_out, int out_size, void* d_ws, size_t ws_size,
                              hipStream_t stream) {
  const float* hidden        = (const float*)d_in[0];
  const float* token_mask    = (const float*)d_in[1];
  const float* pooled_tokens = (const float*)d_in[2];
  const float* w_align       = (const float*)d_in[3];
  const float* b_align       = (const float*)d_in[4];

  float* out        = (float*)d_out;
  float* out_pooled = out;                      // [B,H]
  float* out_probs  = out + (size_t)B_ * H_;    // [B,S,1] — also scores staging
  float* dotA       = (float*)d_ws;             // 64 floats

  k_dot_pooled<<<B_, 256, 0, stream>>>(pooled_tokens, w_align, dotA);
  k_scores_wmma<<<(B_ * S_ / 16) / 8, 256, 0, stream>>>(
      hidden, token_mask, w_align, b_align, dotA, out_probs);
  k_sparsemax<<<B_, 256, 0, stream>>>(token_mask, out_probs);
  k_pool<<<dim3(B_, H_ / 256), 256, 0, stream>>>(hidden, out_probs, out_pooled);
}
